// HyperTemporalTransactionModel_69681549410401
// MI455X (gfx1250) — compile-verified, matrix operand order checked by reading, not run on hardware
//
#include <hip/hip_runtime.h>
#include <math.h>

// ---------------- problem constants ----------------
#define B_    4
#define S_    512
#define H_    512
#define NH_   8
#define T_    3
#define DH_   64
#define DTS_  170
#define EPS_  1e-5f
constexpr int       BS_  = B_ * S_;                     // 2048
constexpr int       H3_  = 3 * H_;                      // 1536
constexpr int       H4_  = 4 * H_;                      // 2048
constexpr long long BSS_ = (long long)B_ * S_ * S_;     // 1048576

// ---------------- CDNA5 WMMA types ----------------
typedef __attribute__((ext_vector_type(16))) __bf16 v16bf;
typedef __attribute__((ext_vector_type(8)))  float  v8f;

union FragB16 {
    uint4          q[2];
    unsigned short u[16];
    v16bf          v;
};

__device__ __forceinline__ unsigned short f2bf(float f) {
    union { float f; unsigned int u; } c; c.f = f;
    unsigned int u = c.u;
    unsigned int r = u + 0x7FFFu + ((u >> 16) & 1u);   // round-to-nearest-even
    return (unsigned short)(r >> 16);
}

__device__ __forceinline__ float gelu_exact(float x) {
    return 0.5f * x * (1.0f + erff(x * 0.70710678118654752f));
}

// CDNA5 async global->LDS copy (16B per lane), tracked by ASYNCcnt.
__device__ __forceinline__ void async_ld_b128(unsigned lds_addr, const void* gaddr) {
    asm volatile("global_load_async_to_lds_b128 %0, %1, off"
                 :: "v"(lds_addr), "v"((unsigned long long)(size_t)gaddr)
                 : "memory");
}
__device__ __forceinline__ void wait_async0() {
    asm volatile("s_wait_asynccnt 0" ::: "memory");
}

// =====================================================================
// fp32 -> bf16 elementwise convert
// =====================================================================
__global__ void cvt_bf16_kernel(const float* __restrict__ src,
                                unsigned short* __restrict__ dst, int n) {
    int i = blockIdx.x * 256 + threadIdx.x;
    if (i < n) dst[i] = f2bf(src[i]);
}

// =====================================================================
// Weight transpose+convert: src fp32 [K][N] -> dst bf16 [N][K]
// K, N multiples of 32. 32x32 LDS tiles, 256 threads.
// =====================================================================
__global__ void twc_kernel(const float* __restrict__ src,
                           unsigned short* __restrict__ dst, int K, int N) {
    __shared__ float t[32][33];
    const int kt = blockIdx.y * 32, nt = blockIdx.x * 32;
    const int lr = threadIdx.x >> 5, lc = threadIdx.x & 31;
    #pragma unroll
    for (int i = 0; i < 4; ++i)
        t[lr + i * 8][lc] = src[(size_t)(kt + lr + i * 8) * N + nt + lc];
    __syncthreads();
    #pragma unroll
    for (int i = 0; i < 4; ++i)
        dst[(size_t)(nt + lr + i * 8) * K + kt + lc] = f2bf(t[lc][lr + i * 8]);
}

// =====================================================================
// Pairwise-time additive attention masks: masks[t][b][q][k]
// =====================================================================
__global__ void mask_kernel(const float* __restrict__ ts,
                            const float* __restrict__ encW,
                            const float* __restrict__ encB,
                            const float* __restrict__ encG,
                            const float* __restrict__ encBeta,
                            float* __restrict__ masks) {
    __shared__ float Wsh[T_][2][DTS_];
    __shared__ float Bsh[T_][DTS_], Gsh[T_][DTS_], Besh[T_][DTS_];
    const int tid = threadIdx.x;
    for (int i = tid; i < T_ * 2 * DTS_; i += 256) ((float*)Wsh)[i] = encW[i];
    for (int i = tid; i < T_ * DTS_; i += 256) {
        ((float*)Bsh)[i]  = encB[i];
        ((float*)Gsh)[i]  = encG[i];
        ((float*)Besh)[i] = encBeta[i];
    }
    __syncthreads();

    const long long idx = (long long)blockIdx.x * 256 + tid;
    if (idx >= BSS_) return;
    const int k = (int)(idx % S_);
    const int q = (int)((idx / S_) % S_);
    const int b = (int)(idx / ((long long)S_ * S_));

    const float td   = ts[b * S_ + q] - ts[b * S_ + k];
    const float dir  = (td > 0.f) ? 1.f : ((td < 0.f) ? -1.f : 0.f);
    const float mag0 = log1pf(fabsf(td) * (1.f / 3600.f));

    float sc = 1.f;
    for (int t = 0; t < T_; ++t) {
        const float mag = mag0 * sc;
        float s = 0.f, s2 = 0.f;
        for (int j = 0; j < DTS_; ++j) {
            float e = dir * Wsh[t][0][j] + mag * Wsh[t][1][j] + Bsh[t][j];
            s += e; s2 += e * e;
        }
        const float mean = s * (1.f / DTS_);
        const float inv  = rsqrtf(s2 * (1.f / DTS_) - mean * mean + EPS_);
        float acc = 0.f;
        for (int j = 0; j < DTS_; ++j) {
            float e = dir * Wsh[t][0][j] + mag * Wsh[t][1][j] + Bsh[t][j];
            acc += gelu_exact((e - mean) * inv * Gsh[t][j] + Besh[t][j]);
        }
        masks[(size_t)t * (size_t)BSS_ + (size_t)idx] = acc * (1.f / DTS_);
        sc *= 0.1f;
    }
}

// =====================================================================
// WMMA GEMM on pre-converted bf16: C[M,N] = act(A[M,K] @ Wt[N,K]^T + bias)
// A: bf16 row-major [M][K]; Wt: bf16 [N][K] (transposed weight).
// 64x64 block tile, BK=32, 8 waves, LDS double-buffered via
// global_load_async_to_lds_b128 (ASYNCcnt). Output fp32 (Cf) or bf16 (Cb).
// =====================================================================
__global__ void gemm_bf16_kernel(const unsigned short* __restrict__ A,
                                 const unsigned short* __restrict__ Wt,
                                 const float* __restrict__ bias,
                                 float* __restrict__ Cf,
                                 unsigned short* __restrict__ Cb,
                                 int M, int N, int K, int act) {
    __shared__ alignas(16) unsigned short As[2][64][32];
    __shared__ alignas(16) unsigned short Bs[2][64][32];
    const int tid  = threadIdx.x;
    const int w    = tid >> 5;
    const int lane = tid & 31;
    const int lrow = lane & 15;
    const int kb   = (lane >> 4) * 8;        // 0 or 8
    const int mblk = blockIdx.y * 64;
    const int nblk = blockIdx.x * 64;
    const int wm   = (w & 3) * 16;
    const int wn   = (w >> 2) * 32;

    // async staging: thread -> one 16B chunk of each tile
    const int cr = tid >> 2;                 // row 0..63
    const int cc = (tid & 3) * 8;            // ushort col 0,8,16,24
    const unsigned short* ag = A  + (size_t)(mblk + cr) * K + cc;
    const unsigned short* bg = Wt + (size_t)(nblk + cr) * K + cc;
    const unsigned asd[2] = { (unsigned)(size_t)&As[0][cr][cc], (unsigned)(size_t)&As[1][cr][cc] };
    const unsigned bsd[2] = { (unsigned)(size_t)&Bs[0][cr][cc], (unsigned)(size_t)&Bs[1][cr][cc] };

    // preload tile 0 into buffer 0
    async_ld_b128(asd[0], ag);
    async_ld_b128(bsd[0], bg);

    v8f acc0 = {}, acc1 = {};
    const int steps = K >> 5;
    for (int s = 0; s < steps; ++s) {
        wait_async0();          // my wave's pending tile writes are in LDS
        __syncthreads();        // everyone's writes visible; prev reads done
        if (s + 1 < steps) {    // overlap next tile's DMA with this tile's math
            const int nxt = (s + 1) & 1;
            async_ld_b128(asd[nxt], ag + (size_t)(s + 1) * 32);
            async_ld_b128(bsd[nxt], bg + (size_t)(s + 1) * 32);
        }
        const int cur = s & 1;
        FragB16 fa, fb0, fb1;
        const uint4* ar = (const uint4*)&As[cur][wm + lrow][0];
        fa.q[0] = ar[kb >> 3];          fa.q[1] = ar[2 + (kb >> 3)];
        const uint4* b0 = (const uint4*)&Bs[cur][wn + lrow][0];
        fb0.q[0] = b0[kb >> 3];         fb0.q[1] = b0[2 + (kb >> 3)];
        const uint4* b1 = (const uint4*)&Bs[cur][wn + 16 + lrow][0];
        fb1.q[0] = b1[kb >> 3];         fb1.q[1] = b1[2 + (kb >> 3)];

        acc0 = __builtin_amdgcn_wmma_f32_16x16x32_bf16(false, fa.v, false, fb0.v,
                                                       (short)0, acc0, false, false);
        acc1 = __builtin_amdgcn_wmma_f32_16x16x32_bf16(false, fa.v, false, fb1.v,
                                                       (short)0, acc1, false, false);
    }

    const int rbase = mblk + wm + ((lane >> 4) << 3);
    const int c0 = nblk + wn + lrow;
    const int c1 = c0 + 16;
    const float bia0 = bias ? bias[c0] : 0.f;
    const float bia1 = bias ? bias[c1] : 0.f;
    #pragma unroll
    for (int r = 0; r < 8; ++r) {
        float v0 = acc0[r] + bia0;
        float v1 = acc1[r] + bia1;
        if (act == 1) { v0 = gelu_exact(v0); v1 = gelu_exact(v1); }
        if (Cb) {
            Cb[(size_t)(rbase + r) * N + c0] = f2bf(v0);
            Cb[(size_t)(rbase + r) * N + c1] = f2bf(v1);
        } else {
            Cf[(size_t)(rbase + r) * N + c0] = v0;
            Cf[(size_t)(rbase + r) * N + c1] = v1;
        }
    }
}

// =====================================================================
// scores[bh][q][k] = scale * Q.K (+ mask[b][q][k]); qkv is bf16 [B,S,3H].
// grid (S/16, B*NH), 256 threads; wave w handles 64 key columns.
// Fragments are direct aligned uint4 global loads (no conversion).
// =====================================================================
__global__ void attn_scores_kernel(const unsigned short* __restrict__ qkvb,
                                   const float* __restrict__ mask,
                                   float* __restrict__ scores,
                                   float scale) {
    const int bh = blockIdx.y;
    const int b  = bh / NH_;
    const int h  = bh % NH_;
    const int q0 = blockIdx.x * 16;
    const int tid = threadIdx.x, w = tid >> 5, lane = tid & 31;
    const int lrow = lane & 15, kb = (lane >> 4) * 8;

    const unsigned short* Qb = qkvb + (size_t)b * S_ * H3_ + h * DH_;
    const unsigned short* Kb = qkvb + (size_t)b * S_ * H3_ + H_ + h * DH_;

    v8f acc[4] = {};
    #pragma unroll
    for (int ks0 = 0; ks0 < DH_; ks0 += 32) {
        FragB16 fa;
        const uint4* qp = (const uint4*)(Qb + (size_t)(q0 + lrow) * H3_ + ks0 + kb);
        fa.q[0] = qp[0]; fa.q[1] = qp[2];
        #pragma unroll
        for (int nt = 0; nt < 4; ++nt) {
            const int key = w * 64 + nt * 16 + lrow;
            const uint4* kp = (const uint4*)(Kb + (size_t)key * H3_ + ks0 + kb);
            FragB16 fb;
            fb.q[0] = kp[0]; fb.q[1] = kp[2];
            acc[nt] = __builtin_amdgcn_wmma_f32_16x16x32_bf16(false, fa.v, false, fb.v,
                                                              (short)0, acc[nt], false, false);
        }
    }

    float* srow = scores + (size_t)bh * S_ * S_;
    const float* mrow = mask ? (mask + (size_t)b * S_ * S_) : nullptr;
    const int rb = q0 + ((lane >> 4) << 3);
    #pragma unroll
    for (int nt = 0; nt < 4; ++nt) {
        const int col = w * 64 + nt * 16 + lrow;
        #pragma unroll
        for (int r = 0; r < 8; ++r) {
            float v = acc[nt][r] * scale;
            if (mrow) v += mrow[(size_t)(rb + r) * S_ + col];
            srow[(size_t)(rb + r) * S_ + col] = v;
        }
    }
}

// =====================================================================
// Row softmax: fp32 scores -> bf16 probs. One wave per row; each lane
// owns 16 contiguous columns (float4 in, packed uint4 bf16 out).
// =====================================================================
__global__ void softmax_kernel(const float* __restrict__ scores,
                               unsigned short* __restrict__ pbf) {
    const int row  = blockIdx.x * 8 + (threadIdx.x >> 5);
    const int lane = threadIdx.x & 31;
    const float4* r4 = (const float4*)(scores + (size_t)row * S_ + lane * 16);
    float vals[16];
    #pragma unroll
    for (int i = 0; i < 4; ++i) {
        float4 v = r4[i];
        vals[i * 4 + 0] = v.x; vals[i * 4 + 1] = v.y;
        vals[i * 4 + 2] = v.z; vals[i * 4 + 3] = v.w;
    }
    float mx = -3.0e38f;
    #pragma unroll
    for (int i = 0; i < 16; ++i) mx = fmaxf(mx, vals[i]);
    for (int m = 16; m; m >>= 1) mx = fmaxf(mx, __shfl_xor(mx, m));
    float sum = 0.f;
    #pragma unroll
    for (int i = 0; i < 16; ++i) { vals[i] = __expf(vals[i] - mx); sum += vals[i]; }
    for (int m = 16; m; m >>= 1) sum += __shfl_xor(sum, m);
    const float inv = 1.f / sum;
    union { uint4 q[2]; unsigned short u[16]; } o;
    #pragma unroll
    for (int i = 0; i < 16; ++i) o.u[i] = f2bf(vals[i] * inv);
    uint4* dst = (uint4*)(pbf + (size_t)row * S_ + lane * 16);
    dst[0] = o.q[0]; dst[1] = o.q[1];
}

// =====================================================================
// O[b,q,h*DH+d] = P[bh,q,:] @ V[b,:,h*DH+d]; P bf16, V bf16 (async->LDS).
// grid (S/16, B*NH), 128 threads (4 waves); wave w owns d-tile w*16.
// Output written directly as bf16 (feeds the out-projection GEMM).
// =====================================================================
__global__ void attn_pv_kernel(const unsigned short* __restrict__ pbf,
                               const unsigned short* __restrict__ qkvb,
                               unsigned short* __restrict__ obf) {
    const int bh = blockIdx.y;
    const int b  = bh / NH_;
    const int h  = bh % NH_;
    const int q0 = blockIdx.x * 16;
    const int tid = threadIdx.x, w = tid >> 5, lane = tid & 31;
    const int lrow = lane & 15, kb = (lane >> 4) * 8;

    __shared__ alignas(16) unsigned short Vs[32][64];   // [key][d] bf16 tile
    const unsigned short* Pr = pbf + (size_t)bh * S_ * S_;
    const unsigned short* Vb = qkvb + (size_t)b * S_ * H3_ + 2 * H_ + h * DH_;

    // async staging: 256 chunks of 16B, 2 per thread
    const int c0i = tid, c1i = tid + 128;
    const unsigned vd0 = (unsigned)(size_t)&Vs[c0i >> 3][(c0i & 7) * 8];
    const unsigned vd1 = (unsigned)(size_t)&Vs[c1i >> 3][(c1i & 7) * 8];

    v8f acc = {};
    for (int k0 = 0; k0 < S_; k0 += 32) {
        __syncthreads();        // prev reads done before overwrite
        async_ld_b128(vd0, Vb + (size_t)(k0 + (c0i >> 3)) * H3_ + (c0i & 7) * 8);
        async_ld_b128(vd1, Vb + (size_t)(k0 + (c1i >> 3)) * H3_ + (c1i & 7) * 8);
        wait_async0();
        __syncthreads();

        FragB16 fa;
        const uint4* pp = (const uint4*)(Pr + (size_t)(q0 + lrow) * S_ + k0 + kb);
        fa.q[0] = pp[0]; fa.q[1] = pp[2];

        FragB16 fb;                                   // B^T fragment: [d][key]
        const int d = w * 16 + lrow;
        #pragma unroll
        for (int j = 0; j < 8; ++j) { fb.u[j] = Vs[kb + j][d]; fb.u[8 + j] = Vs[kb + 16 + j][d]; }

        acc = __builtin_amdgcn_wmma_f32_16x16x32_bf16(false, fa.v, false, fb.v,
                                                      (short)0, acc, false, false);
    }

    const int col = h * DH_ + w * 16 + lrow;
    const int rb  = q0 + ((lane >> 4) << 3);
    #pragma unroll
    for (int r = 0; r < 8; ++r)
        obf[(size_t)((size_t)b * S_ + rb + r) * H_ + col] = f2bf(acc[r]);
}

// =====================================================================
// LayerNorm(+residual)(+GELU); optional bf16 side output for next GEMM.
// =====================================================================
__global__ void row_ln_kernel(const float* __restrict__ x, const float* __restrict__ res,
                              const float* __restrict__ g, const float* __restrict__ beta,
                              float* __restrict__ y, unsigned short* __restrict__ ybf,
                              int act) {
    const int row = blockIdx.x, tid = threadIdx.x;
    const int w = tid >> 5, lane = tid & 31;
    const size_t base = (size_t)row * H_;
    float s = 0.f, s2 = 0.f;
    for (int c = tid; c < H_; c += 256) {
        float v = x[base + c] + (res ? res[base + c] : 0.f);
        s += v; s2 += v * v;
    }
    for (int m = 16; m; m >>= 1) { s += __shfl_xor(s, m); s2 += __shfl_xor(s2, m); }
    __shared__ float ps[8], ps2[8], stats[2];
    if (lane == 0) { ps[w] = s; ps2[w] = s2; }
    __syncthreads();
    if (tid == 0) {
        float ts = 0.f, ts2 = 0.f;
        for (int i = 0; i < 8; ++i) { ts += ps[i]; ts2 += ps2[i]; }
        float mean = ts / H_;
        stats[0] = mean;
        stats[1] = rsqrtf(ts2 / H_ - mean * mean + EPS_);
    }
    __syncthreads();
    const float mean = stats[0], inv = stats[1];
    for (int c = tid; c < H_; c += 256) {
        float v = x[base + c] + (res ? res[base + c] : 0.f);
        float o = (v - mean) * inv * g[c] + beta[c];
        if (act) o = gelu_exact(o);
        if (y)   y[base + c]   = o;
        if (ybf) ybf[base + c] = f2bf(o);
    }
}

// combined_bf[row, t*H + h] = bf16(outs[t][row, h])
__global__ void concat_kernel(const float* __restrict__ outs,
                              unsigned short* __restrict__ combined_bf) {
    const int idx = blockIdx.x * 256 + threadIdx.x;
    if (idx >= BS_ * H3_) return;
    const int hh  = idx % H_;
    const int t   = (idx / H_) % T_;
    const int row = idx / H3_;
    combined_bf[idx] = f2bf(outs[(size_t)t * BS_ * H_ + (size_t)row * H_ + hh]);
}

// mix = softmax(m @ W2 + b2, T=3); weighted = sum_t mix[t]*outs[t] (f32 + bf16)
__global__ void mix_weighted_kernel(const float* __restrict__ m,
                                    const float* __restrict__ W2,
                                    const float* __restrict__ b2,
                                    const float* __restrict__ outs,
                                    float* __restrict__ weighted,
                                    unsigned short* __restrict__ weighted_bf) {
    const int row = blockIdx.x, tid = threadIdx.x;
    const int w = tid >> 5, lane = tid & 31;
    float l0 = 0.f, l1 = 0.f, l2 = 0.f;
    for (int c = tid; c < H_; c += 256) {
        float mv = m[(size_t)row * H_ + c];
        l0 += mv * W2[c * 3 + 0];
        l1 += mv * W2[c * 3 + 1];
        l2 += mv * W2[c * 3 + 2];
    }
    for (int msk = 16; msk; msk >>= 1) {
        l0 += __shfl_xor(l0, msk); l1 += __shfl_xor(l1, msk); l2 += __shfl_xor(l2, msk);
    }
    __shared__ float p[8][3], sm[3];
    if (lane == 0) { p[w][0] = l0; p[w][1] = l1; p[w][2] = l2; }
    __syncthreads();
    if (tid == 0) {
        float t0 = b2[0], t1 = b2[1], t2 = b2[2];
        for (int i = 0; i < 8; ++i) { t0 += p[i][0]; t1 += p[i][1]; t2 += p[i][2]; }
        float mx = fmaxf(t0, fmaxf(t1, t2));
        float e0 = __expf(t0 - mx), e1 = __expf(t1 - mx), e2 = __expf(t2 - mx);
        float inv = 1.f / (e0 + e1 + e2);
        sm[0] = e0 * inv; sm[1] = e1 * inv; sm[2] = e2 * inv;
    }
    __syncthreads();
    for (int c = tid; c < H_; c += 256) {
        const size_t o = (size_t)row * H_ + c;
        float v = sm[0] * outs[o]
                + sm[1] * outs[(size_t)BS_ * H_ + o]
                + sm[2] * outs[(size_t)2 * BS_ * H_ + o];
        weighted[o] = v;
        weighted_bf[o] = f2bf(v);
    }
}

// =====================================================================
extern "C" void kernel_launch(void* const* d_in, const int* in_sizes, int n_in,
                              void* d_out, int out_size, void* d_ws, size_t ws_size,
                              hipStream_t stream) {
    (void)in_sizes; (void)n_in; (void)out_size; (void)ws_size;
    const float* x       = (const float*)d_in[0];
    const float* tstamps = (const float*)d_in[1];
    const float* enc_W   = (const float*)d_in[2];
    const float* enc_b   = (const float*)d_in[3];
    const float* enc_g   = (const float*)d_in[4];
    const float* enc_be  = (const float*)d_in[5];
    const float* qkvW    = (const float*)d_in[6];
    const float* qkvb    = (const float*)d_in[7];
    const float* aoW     = (const float*)d_in[8];
    const float* aob     = (const float*)d_in[9];
    const float* mxW1    = (const float*)d_in[10];
    const float* mxb1    = (const float*)d_in[11];
    const float* mxlng   = (const float*)d_in[12];
    const float* mxlnb   = (const float*)d_in[13];
    const float* mxW2    = (const float*)d_in[14];
    const float* mxb2    = (const float*)d_in[15];
    const float* elqkvW  = (const float*)d_in[16];
    const float* elqkvb  = (const float*)d_in[17];
    const float* eloW    = (const float*)d_in[18];
    const float* elob    = (const float*)d_in[19];
    const float* ln1g    = (const float*)d_in[20];
    const float* ln1b    = (const float*)d_in[21];
    const float* ffW1    = (const float*)d_in[22];
    const float* ffb1    = (const float*)d_in[23];
    const float* ffW2    = (const float*)d_in[24];
    const float* ffb2    = (const float*)d_in[25];
    const float* ln2g    = (const float*)d_in[26];
    const float* ln2b    = (const float*)d_in[27];
    const float* opW     = (const float*)d_in[28];
    const float* opb     = (const float*)d_in[29];
    const float* oplng   = (const float*)d_in[30];
    const float* oplnb   = (const float*)d_in[31];
    float* out = (float*)d_out;

    // ---- workspace layout: fp32 region, then bf16 (ushort) region ----
    float* wsf      = (float*)d_ws;
    float* masks    = wsf;                                   // 3*BSS
    float* scores   = masks    + (size_t)T_ * BSS_;          // B*NH*S*S
    float* outs     = scores   + (size_t)B_ * NH_ * S_ * S_; // 3*BS*H
    float* mbuf     = outs     + (size_t)T_ * BS_ * H_;      // BS*H
    float* weighted = mbuf     + (size_t)BS_ * H_;           // BS*H
    float* sa       = weighted + (size_t)BS_ * H_;           // BS*H
    float* h1       = sa       + (size_t)BS_ * H_;           // BS*H
    float* ff2      = h1       + (size_t)BS_ * H_;           // BS*H
    float* h2       = ff2      + (size_t)BS_ * H_;           // BS*H
    float* opg      = h2       + (size_t)BS_ * H_;           // BS*H

    unsigned short* usb      = (unsigned short*)(opg + (size_t)BS_ * H_);
    unsigned short* xbf      = usb;                              // BS*H
    unsigned short* qkv_bf   = xbf      + (size_t)BS_ * H_;      // BS*3H
    unsigned short* pbf      = qkv_bf   + (size_t)BS_ * H3_;     // B*NH*S*S
    unsigned short* aobf     = pbf      + (size_t)B_ * NH_ * S_ * S_; // BS*H
    unsigned short* comb_bf  = aobf     + (size_t)BS_ * H_;      // BS*3H
    unsigned short* wtd_bf   = comb_bf  + (size_t)BS_ * H3_;     // BS*H
    unsigned short* h1_bf    = wtd_bf   + (size_t)BS_ * H_;      // BS*H
    unsigned short* ff1_bf   = h1_bf    + (size_t)BS_ * H_;      // BS*4H
    unsigned short* h2_bf    = ff1_bf   + (size_t)BS_ * H4_;     // BS*H
    unsigned short* wt_qkv   = h2_bf    + (size_t)BS_ * H_;      // T*H3*H
    unsigned short* wt_ao    = wt_qkv   + (size_t)T_ * H3_ * H_; // T*H*H
    unsigned short* wt_mx1   = wt_ao    + (size_t)T_ * H_ * H_;  // H*H3
    unsigned short* wt_elqkv = wt_mx1   + (size_t)H_ * H3_;      // H3*H
    unsigned short* wt_elo   = wt_elqkv + (size_t)H3_ * H_;      // H*H
    unsigned short* wt_ff1   = wt_elo   + (size_t)H_ * H_;       // H4*H
    unsigned short* wt_ff2   = wt_ff1   + (size_t)H4_ * H_;      // H*H4
    unsigned short* wt_op    = wt_ff2   + (size_t)H_ * H4_;      // H*H

    const float scale = 0.125f;  // 1/sqrt(DH)

    // ---- one-time conversions (weights transposed to [N][K] bf16) ----
    cvt_bf16_kernel<<<(BS_ * H_) / 256, 256, 0, stream>>>(x, xbf, BS_ * H_);
    for (int t = 0; t < T_; ++t) {
        twc_kernel<<<dim3(H3_ / 32, H_ / 32), 256, 0, stream>>>(
            qkvW + (size_t)t * H_ * H3_, wt_qkv + (size_t)t * H3_ * H_, H_, H3_);
        twc_kernel<<<dim3(H_ / 32, H_ / 32), 256, 0, stream>>>(
            aoW + (size_t)t * H_ * H_, wt_ao + (size_t)t * H_ * H_, H_, H_);
    }
    twc_kernel<<<dim3(H_ / 32, H3_ / 32), 256, 0, stream>>>(mxW1, wt_mx1, H3_, H_);
    twc_kernel<<<dim3(H3_ / 32, H_ / 32), 256, 0, stream>>>(elqkvW, wt_elqkv, H_, H3_);
    twc_kernel<<<dim3(H_ / 32, H_ / 32), 256, 0, stream>>>(eloW, wt_elo, H_, H_);
    twc_kernel<<<dim3(H4_ / 32, H_ / 32), 256, 0, stream>>>(ffW1, wt_ff1, H_, H4_);
    twc_kernel<<<dim3(H_ / 32, H4_ / 32), 256, 0, stream>>>(ffW2, wt_ff2, H4_, H_);
    twc_kernel<<<dim3(H_ / 32, H_ / 32), 256, 0, stream>>>(opW, wt_op, H_, H_);

    mask_kernel<<<(int)(BSS_ / 256), 256, 0, stream>>>(tstamps, enc_W, enc_b, enc_g, enc_be, masks);

    // ---- per-timescale attention ----
    for (int t = 0; t < T_; ++t) {
        gemm_bf16_kernel<<<dim3(H3_ / 64, BS_ / 64), 256, 0, stream>>>(
            xbf, wt_qkv + (size_t)t * H3_ * H_, qkvb + t * H3_,
            nullptr, qkv_bf, BS_, H3_, H_, 0);
        attn_scores_kernel<<<dim3(S_ / 16, B_ * NH_), 256, 0, stream>>>(
            qkv_bf, masks + (size_t)t * BSS_, scores, scale);
        softmax_kernel<<<(B_ * NH_ * S_) / 8, 256, 0, stream>>>(scores, pbf);
        attn_pv_kernel<<<dim3(S_ / 16, B_ * NH_), 128, 0, stream>>>(pbf, qkv_bf, aobf);
        gemm_bf16_kernel<<<dim3(H_ / 64, BS_ / 64), 256, 0, stream>>>(
            aobf, wt_ao + (size_t)t * H_ * H_, aob + t * H_,
            outs + (size_t)t * BS_ * H_, nullptr, BS_, H_, H_, 0);
    }

    // ---- time mixer ----
    concat_kernel<<<(BS_ * H3_) / 256, 256, 0, stream>>>(outs, comb_bf);
    gemm_bf16_kernel<<<dim3(H_ / 64, BS_ / 64), 256, 0, stream>>>(
        comb_bf, wt_mx1, mxb1, mbuf, nullptr, BS_, H_, H3_, 0);
    row_ln_kernel<<<BS_, 256, 0, stream>>>(mbuf, nullptr, mxlng, mxlnb, mbuf, nullptr, 1);
    mix_weighted_kernel<<<BS_, 256, 0, stream>>>(mbuf, mxW2, mxb2, outs, weighted, wtd_bf);

    // ---- final post-norm encoder layer ----
    gemm_bf16_kernel<<<dim3(H3_ / 64, BS_ / 64), 256, 0, stream>>>(
        wtd_bf, wt_elqkv, elqkvb, nullptr, qkv_bf, BS_, H3_, H_, 0);
    attn_scores_kernel<<<dim3(S_ / 16, B_ * NH_), 256, 0, stream>>>(qkv_bf, nullptr, scores, scale);
    softmax_kernel<<<(B_ * NH_ * S_) / 8, 256, 0, stream>>>(scores, pbf);
    attn_pv_kernel<<<dim3(S_ / 16, B_ * NH_), 128, 0, stream>>>(pbf, qkv_bf, aobf);
    gemm_bf16_kernel<<<dim3(H_ / 64, BS_ / 64), 256, 0, stream>>>(
        aobf, wt_elo, elob, sa, nullptr, BS_, H_, H_, 0);
    row_ln_kernel<<<BS_, 256, 0, stream>>>(weighted, sa, ln1g, ln1b, h1, h1_bf, 0);

    gemm_bf16_kernel<<<dim3(H4_ / 64, BS_ / 64), 256, 0, stream>>>(
        h1_bf, wt_ff1, ffb1, nullptr, ff1_bf, BS_, H4_, H_, 1);
    gemm_bf16_kernel<<<dim3(H_ / 64, BS_ / 64), 256, 0, stream>>>(
        ff1_bf, wt_ff2, ffb2, ff2, nullptr, BS_, H_, H4_, 0);
    row_ln_kernel<<<BS_, 256, 0, stream>>>(h1, ff2, ln2g, ln2b, h2, h2_bf, 0);

    gemm_bf16_kernel<<<dim3(H_ / 64, BS_ / 64), 256, 0, stream>>>(
        h2_bf, wt_op, opb, opg, nullptr, BS_, H_, H_, 0);
    row_ln_kernel<<<BS_, 256, 0, stream>>>(opg, nullptr, oplng, oplnb, out, nullptr, 0);
}
